// SchNet_60498909331542
// MI455X (gfx1250) — matrix operand order repaired and checked by reading, not compiled, not run on hardware
//
#include <hip/hip_runtime.h>
#include <hip/hip_bf16.h>

#define N_ATOMS 50000
#define N_PAD   50048            // 391 * 128, tile-aligned node count
#define E_EDGES 1600000
#define G_GRAPHS 250
#define HC 64
#define NG 50
#define NB 3
#define CUTOFF_R 10.0f

typedef _Float16 half_t;
typedef half_t v16h __attribute__((ext_vector_type(16)));
typedef float  v8f  __attribute__((ext_vector_type(8)));
typedef unsigned int u32v4 __attribute__((ext_vector_type(4)));
typedef int i32v8 __attribute__((ext_vector_type(8)));
typedef int i32v4 __attribute__((ext_vector_type(4)));

__device__ __forceinline__ float sspf(float v) {
    // softplus(v) - ln2, numerically stable
    return fmaxf(v, 0.0f) + log1pf(expf(-fabsf(v))) - 0.69314718055994530942f;
}

// ---- WMMA fragment loaders (16-bit, wave32 layouts per CDNA5 ISA 7.12.2) ----
__device__ __forceinline__ v16h lds_a_frag(const half_t* tile, int kstart) {
    int lane = threadIdx.x & 31;
    const half_t* p = tile + (lane & 15) * 72 + kstart + ((lane < 16) ? 0 : 8);
    v16h a;
#pragma unroll
    for (int i = 0; i < 8; ++i) { a[i] = p[i]; a[i + 8] = p[16 + i]; }
    return a;
}
__device__ __forceinline__ v16h lds_b_frag(const half_t* wt, int ncol, int kstart) {
    int lane = threadIdx.x & 31;
    const half_t* p = wt + ncol * 72 + kstart + ((lane < 16) ? 0 : 8);
    v16h b;
#pragma unroll
    for (int i = 0; i < 8; ++i) { b[i] = p[i]; b[i + 8] = p[16 + i]; }
    return b;
}
__device__ __forceinline__ v8f wmma_k64(v16h a0, v16h a1, const half_t* wt, int nt, v8f acc) {
    int n = nt * 16 + (threadIdx.x & 15);
    v16h b0 = lds_b_frag(wt, n, 0);
    v16h b1 = lds_b_frag(wt, n, 32);
    acc = __builtin_amdgcn_wmma_f32_16x16x32_f16(false, a0, false, b0, (short)0, acc, false, false);
    acc = __builtin_amdgcn_wmma_f32_16x16x32_f16(false, a1, false, b1, (short)0, acc, false, false);
    return acc;
}
// 16x64 @ 64x64 with per-column f32 bias: 4 independent accumulator chains
// (keeps WMMA pipe busy; defers accumulator consumption past all 8 WMMAs)
__device__ __forceinline__ void gemm_all(v16h a0, v16h a1, const half_t* wt,
                                         const float* bias, v8f accs[4]) {
    int nl = threadIdx.x & 15;
#pragma unroll
    for (int nt = 0; nt < 4; ++nt) {
        float bb = bias ? bias[nt * 16 + nl] : 0.0f;
        v8f c;
#pragma unroll
        for (int r = 0; r < 8; ++r) c[r] = bb;
        accs[nt] = wmma_k64(a0, a1, wt, nt, c);
    }
}

// ---- TDM: load one 64x64 f16 weight tile into LDS with 72-half row stride ----
// pad_enable: after each 32 dwords (one 64-half row) insert 4 dwords pad -> 144B stride.
__device__ __forceinline__ void tdm_load_weight(const half_t* gsrc, half_t* ldst) {
#if __has_builtin(__builtin_amdgcn_tensor_load_to_lds)
    unsigned long long ga = (unsigned long long)(uintptr_t)gsrc;
    unsigned int ldsoff =
        (unsigned int)__builtin_amdgcn_readfirstlane((int)(uintptr_t)ldst);
    u32v4 g0;
    g0[0] = 1u;                                        // count=1, user descriptor
    g0[1] = ldsoff;                                    // lds_addr
    g0[2] = (unsigned int)(ga & 0xffffffffull);        // global_addr[31:0]
    g0[3] = ((unsigned int)(ga >> 32) & 0x01ffffffu) | (2u << 30);  // addr[56:32] | type=2
    i32v8 g1;
    g1[0] = (1 << 16) | (1 << 20) | (4 << 22) | (3 << 25); // data_size=2B, pad_en, 32dw, pad 4dw
    g1[1] = (64 << 16);                                // tensor_dim0 = 64
    g1[2] = (64 << 16);                                // tensor_dim1 = 64
    g1[3] = (64 << 16);                                // tile_dim0 = 64
    g1[4] = 64;                                        // tile_dim1 = 64, tile_dim2 = 0
    g1[5] = 64;                                        // tensor_dim0_stride = 64
    g1[6] = 0; g1[7] = 0;
    i32v4 zz = {0, 0, 0, 0};
#if defined(__clang_major__) && (__clang_major__ >= 23)
    i32v8 zz8 = {0, 0, 0, 0, 0, 0, 0, 0};
    __builtin_amdgcn_tensor_load_to_lds(g0, g1, zz, zz, zz8, 0);
#else
    __builtin_amdgcn_tensor_load_to_lds(g0, g1, zz, zz, 0);
#endif
#else
    int lane = threadIdx.x & 31;
    for (int i = lane; i < 4096; i += 32)
        ldst[(i >> 6) * 72 + (i & 63)] = gsrc[i];
#endif
}
__device__ __forceinline__ void tdm_wait() {
#if __has_builtin(__builtin_amdgcn_s_wait_tensorcnt)
    __builtin_amdgcn_s_wait_tensorcnt(0);
#endif
}

// ---------------- small utility kernels ----------------
__global__ void k_fill(float* p, int n) {
    int i = blockIdx.x * 256 + threadIdx.x;
    if (i < n) p[i] = 0.0f;
}

__global__ void k_embed(const int* __restrict__ z, const float* __restrict__ emb,
                        float* __restrict__ h) {
    int i = blockIdx.x * 256 + threadIdx.x;   // over N_PAD*HC
    if (i < N_PAD * HC) {
        int a = i >> 6, f = i & 63;
        h[i] = (a < N_ATOMS) ? emb[z[a] * HC + f] : 0.0f;
    }
}

__global__ void k_geom(const int* __restrict__ ei, const float* __restrict__ pos,
                       float* __restrict__ dist, float* __restrict__ cc) {
    int e = blockIdx.x * 256 + threadIdx.x;
    if (e < E_EDGES) {
        int s = ei[e], t = ei[E_EDGES + e];
        float dx = pos[t * 3 + 0] - pos[s * 3 + 0];
        float dy = pos[t * 3 + 1] - pos[s * 3 + 1];
        float dz = pos[t * 3 + 2] - pos[s * 3 + 2];
        float dd = sqrtf(dx * dx + dy * dy + dz * dz);
        dist[e] = dd;
        cc[e] = 0.5f * (cosf(dd * (3.14159265358979323846f / CUTOFF_R)) + 1.0f);
    }
}

// pack fp32 weight [NB][K][64] -> f16 [NB][64(n)][64(k)], zero-pad K to 64
__global__ void k_pack(const float* __restrict__ src, half_t* __restrict__ dst, int K) {
    int i = blockIdx.x * 256 + threadIdx.x;
    if (i < NB * 64 * 64) {
        int t = i >> 12, r = i & 4095, n = r >> 6, k = r & 63;
        float v = (k < K) ? src[t * K * 64 + k * 64 + n] : 0.0f;
        dst[i] = (half_t)v;
    }
}

// ---------------- node GEMM: hw = h @ cf1_w[t]  (f16 out, padded rows) ----------------
__global__ void __launch_bounds__(256) k_node_gemm(const float* __restrict__ x,
                                                   const half_t* __restrict__ wglob,
                                                   half_t* __restrict__ out) {
    __shared__ __attribute__((aligned(16))) half_t ws[64 * 72];
    __shared__ __attribute__((aligned(16))) half_t at[8][16 * 72];
    int tid = threadIdx.x, wv = tid >> 5, lane = tid & 31;
    if (wv == 0) { tdm_load_weight(wglob, ws); tdm_wait(); }
    int base = blockIdx.x * 128 + wv * 16;
    int m = lane & 15, co = (lane < 16) ? 0 : 32;
    const float* xp = x + (size_t)(base + m) * HC + co;
#pragma unroll
    for (int j = 0; j < 32; ++j) at[wv][m * 72 + co + j] = (half_t)xp[j];
    __syncthreads();
    v16h a0 = lds_a_frag(at[wv], 0);
    v16h a1 = lds_a_frag(at[wv], 32);
    v8f accs[4];
    gemm_all(a0, a1, ws, nullptr, accs);
    int g8 = (lane < 16) ? 0 : 8;
#pragma unroll
    for (int nt = 0; nt < 4; ++nt) {
        int n = nt * 16 + (lane & 15);
#pragma unroll
        for (int r = 0; r < 8; ++r)
            at[wv][(r + g8) * 72 + n] = (half_t)accs[nt][r];  // stage back into A-tile
    }
    __syncthreads();
    // coalesced row-major writeout: 64B per lane
    const half_t* sp = &at[wv][m * 72 + co];
    half_t* dp = out + (size_t)(base + m) * HC + co;
#pragma unroll
    for (int j = 0; j < 32; ++j) dp[j] = sp[j];
}

// ---------------- fused edge kernel: RBF -> MLP(WMMA) -> modulate -> scatter ----------------
__global__ void __launch_bounds__(256) k_edge(const int* __restrict__ ei,
                                              const float* __restrict__ dist,
                                              const float* __restrict__ cc,
                                              const half_t* __restrict__ w1g,
                                              const float* __restrict__ b1,
                                              const half_t* __restrict__ w2g,
                                              const float* __restrict__ b2,
                                              const half_t* __restrict__ hw,
                                              float* __restrict__ agg) {
    __shared__ __attribute__((aligned(16))) half_t w1s[64 * 72];
    __shared__ __attribute__((aligned(16))) half_t w2s[64 * 72];
    __shared__ __attribute__((aligned(16))) half_t rb[8][16 * 72];
    __shared__ int   sI[8][16];
    __shared__ int   dI[8][16];
    __shared__ float Cw[8][16];

    int tid = threadIdx.x, wv = tid >> 5, lane = tid & 31;
    if (wv == 0) { tdm_load_weight(w1g, w1s); tdm_load_weight(w2g, w2s); tdm_wait(); }
    int e0 = blockIdx.x * 128 + wv * 16;       // E divisible by 128: no guard
    int m = lane & 15, ko = (lane < 16) ? 0 : 32;
    int e = e0 + m;
    float dd = dist[e];
    if (lane < 16) {
        sI[wv][lane] = ei[e];
        dI[wv][lane] = ei[E_EDGES + e];
        Cw[wv][lane] = cc[e];
    }
    const float delta = CUTOFF_R / (float)(NG - 1);
    const float coeff = -0.5f / (delta * delta);
#pragma unroll
    for (int j = 0; j < 32; ++j) {
        int k = ko + j;
        float xx = dd - (float)k * delta;
        float r = (k < NG) ? expf(coeff * xx * xx) : 0.0f;
        rb[wv][m * 72 + k] = (half_t)r;
    }
    __syncthreads();

    int g8 = (lane < 16) ? 0 : 8;
    // GEMM1: [16,64(rbf pad)] x [64,64] + b1, shifted-softplus, staged back into rb
    {
        v16h a0 = lds_a_frag(rb[wv], 0);
        v16h a1 = lds_a_frag(rb[wv], 32);
        v8f accs[4];
        gemm_all(a0, a1, w1s, b1, accs);
#pragma unroll
        for (int nt = 0; nt < 4; ++nt) {
            int n = nt * 16 + (lane & 15);
#pragma unroll
            for (int r = 0; r < 8; ++r)
                rb[wv][(r + g8) * 72 + n] = (half_t)sspf(accs[nt][r]);
        }
    }
    __syncthreads();

    // GEMM2: [16,64] x [64,64] + b2, * cutoff, * hw[src], atomic scatter to agg[dst]
    {
        v16h a0 = lds_a_frag(rb[wv], 0);
        v16h a1 = lds_a_frag(rb[wv], 32);
        v8f accs[4];
        gemm_all(a0, a1, w2s, b2, accs);
        float cv[8]; int si[8], di[8];
#pragma unroll
        for (int r = 0; r < 8; ++r) {
            cv[r] = Cw[wv][r + g8]; si[r] = sI[wv][r + g8]; di[r] = dI[wv][r + g8];
        }
#pragma unroll
        for (int nt = 0; nt < 4; ++nt) {
            int n = nt * 16 + (lane & 15);
#pragma unroll
            for (int r = 0; r < 8; ++r) {
                float wf = accs[nt][r] * cv[r];
                float hv = (float)hw[(size_t)si[r] * HC + n];
                unsafeAtomicAdd(&agg[(size_t)di[r] * HC + n], wf * hv);
            }
        }
    }
}

// ---------------- fused node update: h += ssp(agg@cf2+b)@lin+b (padded rows) ----------------
__global__ void __launch_bounds__(256) k_node_update(const float* __restrict__ agg,
                                                     const half_t* __restrict__ w2g,
                                                     const float* __restrict__ b2,
                                                     const half_t* __restrict__ w3g,
                                                     const float* __restrict__ b3,
                                                     float* __restrict__ h) {
    __shared__ __attribute__((aligned(16))) half_t w2s[64 * 72];
    __shared__ __attribute__((aligned(16))) half_t w3s[64 * 72];
    __shared__ __attribute__((aligned(16))) half_t at[8][16 * 72];
    __shared__ __attribute__((aligned(16))) half_t act[8][16 * 72];
    int tid = threadIdx.x, wv = tid >> 5, lane = tid & 31;
    if (wv == 0) { tdm_load_weight(w2g, w2s); tdm_load_weight(w3g, w3s); tdm_wait(); }
    int base = blockIdx.x * 128 + wv * 16;
    int m = lane & 15, co = (lane < 16) ? 0 : 32;
    int g8 = (lane < 16) ? 0 : 8;
    const float* xp = agg + (size_t)(base + m) * HC + co;
#pragma unroll
    for (int j = 0; j < 32; ++j) at[wv][m * 72 + co + j] = (half_t)xp[j];
    __syncthreads();
    {
        v16h a0 = lds_a_frag(at[wv], 0);
        v16h a1 = lds_a_frag(at[wv], 32);
        v8f accs[4];
        gemm_all(a0, a1, w2s, b2, accs);
#pragma unroll
        for (int nt = 0; nt < 4; ++nt) {
            int n = nt * 16 + (lane & 15);
#pragma unroll
            for (int r = 0; r < 8; ++r)
                act[wv][(r + g8) * 72 + n] = (half_t)sspf(accs[nt][r]);
        }
    }
    __syncthreads();
    {
        v16h a0 = lds_a_frag(act[wv], 0);
        v16h a1 = lds_a_frag(act[wv], 32);
        v8f accs[4];
        gemm_all(a0, a1, w3s, b3, accs);
#pragma unroll
        for (int nt = 0; nt < 4; ++nt) {
            int n = nt * 16 + (lane & 15);
#pragma unroll
            for (int r = 0; r < 8; ++r) {   // branchless residual RMW (rows padded)
                int rr = base + r + g8;
                h[(size_t)rr * HC + n] += accs[nt][r];
            }
        }
    }
}

// ---------------- readout ----------------
__global__ void k_pool(const float* __restrict__ h, const int* __restrict__ batch,
                       float* __restrict__ g) {
    int i = blockIdx.x * 256 + threadIdx.x;
    if (i < N_ATOMS * HC) {
        int a = i >> 6, f = i & 63;
        unsafeAtomicAdd(&g[batch[a] * HC + f], h[i]);
    }
}

__global__ void k_head(const float* __restrict__ g, const float* __restrict__ w1,
                       const float* __restrict__ b1, const float* __restrict__ w2,
                       const float* __restrict__ b2, float* __restrict__ out) {
    int gi = blockIdx.x;
    int j = threadIdx.x;          // 32 threads = 1 wave
    float acc = b1[j];
    for (int i = 0; i < HC; ++i) acc += g[gi * HC + i] * w1[i * 32 + j];
    acc = fmaxf(acc, 0.0f) * w2[j];
#pragma unroll
    for (int off = 16; off > 0; off >>= 1) acc += __shfl_down(acc, off, 32);
    if (j == 0) out[gi] = acc + b2[0];
}

extern "C" void kernel_launch(void* const* d_in, const int* in_sizes, int n_in,
                              void* d_out, int out_size, void* d_ws, size_t ws_size,
                              hipStream_t stream) {
    const int*   z      = (const int*)d_in[0];
    const float* pos    = (const float*)d_in[1];
    const int*   batch  = (const int*)d_in[2];
    const int*   ei     = (const int*)d_in[3];
    const float* emb    = (const float*)d_in[4];
    const float* mlp_w1 = (const float*)d_in[5];
    const float* mlp_b1 = (const float*)d_in[6];
    const float* mlp_w2 = (const float*)d_in[7];
    const float* mlp_b2 = (const float*)d_in[8];
    const float* cf1_w  = (const float*)d_in[9];
    const float* cf2_w  = (const float*)d_in[10];
    const float* cf2_b  = (const float*)d_in[11];
    const float* lin_w  = (const float*)d_in[12];
    const float* lin_b  = (const float*)d_in[13];
    const float* out1_w = (const float*)d_in[14];
    const float* out1_b = (const float*)d_in[15];
    const float* out2_w = (const float*)d_in[16];
    const float* out2_b = (const float*)d_in[17];
    float* out = (float*)d_out;

    // workspace carve-up (node buffers padded to N_PAD rows)
    char* p = (char*)d_ws;
    float* h    = (float*)p; p += (size_t)N_PAD * HC * 4;
    float* agg  = (float*)p; p += (size_t)N_PAD * HC * 4;
    float* dist = (float*)p; p += (size_t)E_EDGES * 4;
    float* ccut = (float*)p; p += (size_t)E_EDGES * 4;
    float* g    = (float*)p; p += (size_t)G_GRAPHS * HC * 4;
    half_t* hw   = (half_t*)p; p += (size_t)N_PAD * HC * 2;
    half_t* w1p  = (half_t*)p; p += (size_t)NB * 64 * 64 * 2;
    half_t* w2p  = (half_t*)p; p += (size_t)NB * 64 * 64 * 2;
    half_t* cf1p = (half_t*)p; p += (size_t)NB * 64 * 64 * 2;
    half_t* cf2p = (half_t*)p; p += (size_t)NB * 64 * 64 * 2;
    half_t* linp = (half_t*)p; p += (size_t)NB * 64 * 64 * 2;

    // preprocessing
    k_embed<<<(N_PAD * HC) / 256, 256, 0, stream>>>(z, emb, h);
    k_geom<<<(E_EDGES + 255) / 256, 256, 0, stream>>>(ei, pos, dist, ccut);
    k_pack<<<48, 256, 0, stream>>>(mlp_w1, w1p, NG);
    k_pack<<<48, 256, 0, stream>>>(mlp_w2, w2p, 64);
    k_pack<<<48, 256, 0, stream>>>(cf1_w, cf1p, 64);
    k_pack<<<48, 256, 0, stream>>>(cf2_w, cf2p, 64);
    k_pack<<<48, 256, 0, stream>>>(lin_w, linp, 64);

    const int nodeBlocks = N_PAD / 128;   // 391 exact
    const int edgeBlocks = E_EDGES / 128; // 12500 exact

    for (int t = 0; t < NB; ++t) {
        k_node_gemm<<<nodeBlocks, 256, 0, stream>>>(h, cf1p + (size_t)t * 4096, hw);
        k_fill<<<(N_PAD * HC) / 256, 256, 0, stream>>>(agg, N_PAD * HC);
        k_edge<<<edgeBlocks, 256, 0, stream>>>(ei, dist, ccut,
                                               w1p + (size_t)t * 4096, mlp_b1 + t * 64,
                                               w2p + (size_t)t * 4096, mlp_b2 + t * 64,
                                               hw, agg);
        k_node_update<<<nodeBlocks, 256, 0, stream>>>(agg,
                                                      cf2p + (size_t)t * 4096, cf2_b + t * 64,
                                                      linp + (size_t)t * 4096, lin_b + t * 64,
                                                      h);
    }

    k_fill<<<(G_GRAPHS * HC + 255) / 256, 256, 0, stream>>>(g, G_GRAPHS * HC);
    k_pool<<<(N_ATOMS * HC + 255) / 256, 256, 0, stream>>>(h, batch, g);
    k_head<<<G_GRAPHS, 32, 0, stream>>>(g, out1_w, out1_b, out2_w, out2_b, out);
}
